// attention_79989470921190
// MI455X (gfx1250) — matrix operand order repaired and checked by reading, not compiled, not run on hardware
//
#include <hip/hip_runtime.h>
#include <hip/hip_bf16.h>

typedef _Float16 v16h __attribute__((ext_vector_type(16)));
typedef _Float16 v8h  __attribute__((ext_vector_type(8)));
typedef float    v8f  __attribute__((ext_vector_type(8)));

#define HEADS 7
#define HD    64
#define SEQ   2048
#define BAT   4
#define HID   896
#define PROJ  448

union AFrag { v16h v; v8h h[2]; };

static __device__ inline v8f wmma_f16(v16h a, v16h b, v8f c) {
  // D = A(16x32) * B(32x16) + C, f32 accumulate
  return __builtin_amdgcn_wmma_f32_16x16x32_f16(false, a, false, b, (short)0, c, false, false);
}

__global__ __launch_bounds__(256) void cvt_f32_f16(const float* __restrict__ src,
                                                   _Float16* __restrict__ dst, int n) {
  int i = blockIdx.x * blockDim.x + threadIdx.x;
  int stride = gridDim.x * blockDim.x;
  for (; i < n; i += stride) dst[i] = (_Float16)src[i];
}

// ---------------------------------------------------------------------------
// QKV projection + bias + RoPE. z=0:q (scaled 1/8), z=1:k, z=2:v (stored
// transposed (b,h,d,s)). Each wave: 32 rows of M, one head's 64 columns.
// Double-buffered A/B fragment prefetch.
// ---------------------------------------------------------------------------
__global__ __launch_bounds__(128) void qkv_rope_kernel(
    const _Float16* __restrict__ xh,
    const _Float16* __restrict__ qwh, const _Float16* __restrict__ kwh,
    const _Float16* __restrict__ vwh,
    const float* __restrict__ qbias, const float* __restrict__ kbias,
    const float* __restrict__ vbias,
    const float* __restrict__ cosp, const float* __restrict__ sinp,
    _Float16* __restrict__ qh, _Float16* __restrict__ kh,
    _Float16* __restrict__ vth)
{
  const int z    = blockIdx.z;
  const _Float16* w   = (z == 0) ? qwh : (z == 1) ? kwh : vwh;
  const float*    bia = (z == 0) ? qbias : (z == 1) ? kbias : vbias;
  const int head = blockIdx.y;
  const int wv   = threadIdx.x >> 5;
  const int lane = threadIdx.x & 31;
  const int g    = lane >> 4, ln = lane & 15;
  const int mbase = blockIdx.x * 128 + wv * 32;

  v8f acc[2][4] = {};
  const _Float16* arow0 = xh + (size_t)(mbase + ln) * HID;
  const _Float16* arow1 = xh + (size_t)(mbase + 16 + ln) * HID;
  const _Float16* wrow  = w + (size_t)(head * HD + ln) * HID;

  AFrag a0, a1; v16h b0, b1, b2, b3;
  a0.h[0] = *(const v8h*)(arow0 + 16 * g);
  a0.h[1] = *(const v8h*)(arow0 + 8 + 16 * g);
  a1.h[0] = *(const v8h*)(arow1 + 16 * g);
  a1.h[1] = *(const v8h*)(arow1 + 8 + 16 * g);
  b0 = *(const v16h*)(wrow + (size_t)0 * 16 * HID + 16 * g);
  b1 = *(const v16h*)(wrow + (size_t)1 * 16 * HID + 16 * g);
  b2 = *(const v16h*)(wrow + (size_t)2 * 16 * HID + 16 * g);
  b3 = *(const v16h*)(wrow + (size_t)3 * 16 * HID + 16 * g);

  for (int kk = 0; kk < HID; kk += 32) {
    const int kn = (kk + 32 < HID) ? kk + 32 : kk;   // branchless tail
    AFrag na0, na1; v16h nb0, nb1, nb2, nb3;
    na0.h[0] = *(const v8h*)(arow0 + kn + 16 * g);
    na0.h[1] = *(const v8h*)(arow0 + kn + 8 + 16 * g);
    na1.h[0] = *(const v8h*)(arow1 + kn + 16 * g);
    na1.h[1] = *(const v8h*)(arow1 + kn + 8 + 16 * g);
    nb0 = *(const v16h*)(wrow + (size_t)0 * 16 * HID + kn + 16 * g);
    nb1 = *(const v16h*)(wrow + (size_t)1 * 16 * HID + kn + 16 * g);
    nb2 = *(const v16h*)(wrow + (size_t)2 * 16 * HID + kn + 16 * g);
    nb3 = *(const v16h*)(wrow + (size_t)3 * 16 * HID + kn + 16 * g);

    acc[0][0] = wmma_f16(a0.v, b0, acc[0][0]);
    acc[0][1] = wmma_f16(a0.v, b1, acc[0][1]);
    acc[0][2] = wmma_f16(a0.v, b2, acc[0][2]);
    acc[0][3] = wmma_f16(a0.v, b3, acc[0][3]);
    acc[1][0] = wmma_f16(a1.v, b0, acc[1][0]);
    acc[1][1] = wmma_f16(a1.v, b1, acc[1][1]);
    acc[1][2] = wmma_f16(a1.v, b2, acc[1][2]);
    acc[1][3] = wmma_f16(a1.v, b3, acc[1][3]);

    a0 = na0; a1 = na1; b0 = nb0; b1 = nb1; b2 = nb2; b3 = nb3;
  }

  float vals[2][4][8];
#pragma unroll
  for (int rr = 0; rr < 2; ++rr)
#pragma unroll
    for (int t = 0; t < 4; ++t) {
      float bv = bia[head * HD + t * 16 + ln];
#pragma unroll
      for (int r = 0; r < 8; ++r) vals[rr][t][r] = acc[rr][t][r] + bv;
    }

  if (z == 2) {
#pragma unroll
    for (int rr = 0; rr < 2; ++rr)
#pragma unroll
      for (int t = 0; t < 4; ++t) {
        int d = t * 16 + ln;
#pragma unroll
        for (int r = 0; r < 8; ++r) {
          int m = mbase + rr * 16 + r + 8 * g;
          int b = m >> 11, s = m & (SEQ - 1);
          vth[((size_t)(b * HEADS + head) * HD + d) * SEQ + s] = (_Float16)vals[rr][t][r];
        }
      }
  } else {
    _Float16* dst = (z == 0) ? qh : kh;
    const float sc = (z == 0) ? 0.125f : 1.0f;   // fold 1/sqrt(64) into q
#pragma unroll
    for (int rr = 0; rr < 2; ++rr)
#pragma unroll
      for (int t = 0; t < 4; ++t) {
        int d = t * 16 + ln;
#pragma unroll
        for (int r = 0; r < 8; ++r) {
          int m = mbase + rr * 16 + r + 8 * g;
          int b = m >> 11, s = m & (SEQ - 1);
          float cs  = cosp[s * HD + d];
          float sn  = sinp[s * HD + d];
          float rot = (t < 2) ? -vals[rr][t + 2][r] : vals[rr][t - 2][r];  // rotate_half
          float res = (vals[rr][t][r] * cs + rot * sn) * sc;
          dst[((size_t)(b * HEADS + head) * SEQ + s) * HD + d] = (_Float16)res;
        }
      }
  }
}

// ---------------------------------------------------------------------------
// Flash attention, transposed formulation: S^T = K * Q^T  (M=keys, N=queries)
// so the softmax reduction over keys is lane-local + one shfl_xor(16).
// Out^T = V^T * P^T accumulated over all 2048 keys. One wave = 16 queries.
// K-fragments for the next key block and V-fragments for the current block
// are prefetched before the softmax VALU work.
// ---------------------------------------------------------------------------
__global__ __launch_bounds__(128) void attn_kernel(
    const _Float16* __restrict__ qh, const _Float16* __restrict__ kh,
    const _Float16* __restrict__ vth, _Float16* __restrict__ ao)
{
  const int bh   = blockIdx.y;
  const int wv   = threadIdx.x >> 5;
  const int lane = threadIdx.x & 31;
  const int g    = lane >> 4, ln = lane & 15;
  const int qbase = (blockIdx.x * 4 + wv) * 16;

  const _Float16* qp = qh + (size_t)bh * SEQ * HD;
  const _Float16* kp = kh + (size_t)bh * SEQ * HD;
  const _Float16* vp = vth + (size_t)bh * HD * SEQ;

  // Q as B-operand: lane ln = query column, contiguous 16 d values
  v16h qb0 = *(const v16h*)(qp + (size_t)(qbase + ln) * HD + 16 * g);
  v16h qb1 = *(const v16h*)(qp + (size_t)(qbase + ln) * HD + 32 + 16 * g);

  float mrun = -1e30f, ssum = 0.0f;
  v8f o[4] = {};

  AFrag a00, a01, a10, a11;
  {
    const _Float16* kr0 = kp + (size_t)ln * HD;
    const _Float16* kr1 = kp + (size_t)(16 + ln) * HD;
    a00.h[0] = *(const v8h*)(kr0 + 16 * g);      a00.h[1] = *(const v8h*)(kr0 + 8 + 16 * g);
    a01.h[0] = *(const v8h*)(kr0 + 32 + 16 * g); a01.h[1] = *(const v8h*)(kr0 + 40 + 16 * g);
    a10.h[0] = *(const v8h*)(kr1 + 16 * g);      a10.h[1] = *(const v8h*)(kr1 + 8 + 16 * g);
    a11.h[0] = *(const v8h*)(kr1 + 32 + 16 * g); a11.h[1] = *(const v8h*)(kr1 + 40 + 16 * g);
  }

  for (int kb = 0; kb < SEQ; kb += 32) {
    // prefetch next key block's K fragments (branchless tail: reload self)
    const int kbn = (kb + 32 < SEQ) ? kb + 32 : kb;
    AFrag n00, n01, n10, n11;
    {
      const _Float16* kr0 = kp + (size_t)(kbn + ln) * HD;
      const _Float16* kr1 = kp + (size_t)(kbn + 16 + ln) * HD;
      n00.h[0] = *(const v8h*)(kr0 + 16 * g);      n00.h[1] = *(const v8h*)(kr0 + 8 + 16 * g);
      n01.h[0] = *(const v8h*)(kr0 + 32 + 16 * g); n01.h[1] = *(const v8h*)(kr0 + 40 + 16 * g);
      n10.h[0] = *(const v8h*)(kr1 + 16 * g);      n10.h[1] = *(const v8h*)(kr1 + 8 + 16 * g);
      n11.h[0] = *(const v8h*)(kr1 + 32 + 16 * g); n11.h[1] = *(const v8h*)(kr1 + 40 + 16 * g);
    }
    // prefetch current V^T fragments (consumed after softmax)
    AFrag va0, va1, va2, va3;
    {
      const _Float16* vr0 = vp + (size_t)(0 * 16 + ln) * SEQ + kb;
      const _Float16* vr1 = vp + (size_t)(1 * 16 + ln) * SEQ + kb;
      const _Float16* vr2 = vp + (size_t)(2 * 16 + ln) * SEQ + kb;
      const _Float16* vr3 = vp + (size_t)(3 * 16 + ln) * SEQ + kb;
      va0.h[0] = *(const v8h*)(vr0 + 16 * g); va0.h[1] = *(const v8h*)(vr0 + 8 + 16 * g);
      va1.h[0] = *(const v8h*)(vr1 + 16 * g); va1.h[1] = *(const v8h*)(vr1 + 8 + 16 * g);
      va2.h[0] = *(const v8h*)(vr2 + 16 * g); va2.h[1] = *(const v8h*)(vr2 + 8 + 16 * g);
      va3.h[0] = *(const v8h*)(vr3 + 16 * g); va3.h[1] = *(const v8h*)(vr3 + 8 + 16 * g);
    }

    v8f s0 = {}, s1 = {};
    s0 = wmma_f16(a00.v, qb0, s0); s0 = wmma_f16(a01.v, qb1, s0);
    s1 = wmma_f16(a10.v, qb0, s1); s1 = wmma_f16(a11.v, qb1, s1);

    // online softmax: per-query max over this lane's 16 keys + xor-16
    float mloc = -1e30f;
#pragma unroll
    for (int r = 0; r < 8; ++r) { mloc = fmaxf(mloc, s0[r]); mloc = fmaxf(mloc, s1[r]); }
    mloc = fmaxf(mloc, __shfl_xor(mloc, 16));
    float mnew = fmaxf(mrun, mloc);
    float corr = __expf(mrun - mnew);

    float p0[8], p1[8], ps = 0.0f;
#pragma unroll
    for (int r = 0; r < 8; ++r) {
      p0[r] = __expf(s0[r] - mnew);
      p1[r] = __expf(s1[r] - mnew);
      ps += p0[r] + p1[r];
    }
    ps += __shfl_xor(ps, 16);
    ssum = ssum * corr + ps;
    mrun = mnew;
#pragma unroll
    for (int t = 0; t < 4; ++t)
#pragma unroll
      for (int r = 0; r < 8; ++r) o[t][r] = o[t][r] * corr;

    // Repack P^T C-layout -> B-layout (K=32 keys) with xor-16 exchanges
    v16h pb;
#pragma unroll
    for (int r = 0; r < 8; ++r) {
      float t0 = (g == 0) ? p1[r] : p0[r];
      float u  = __shfl_xor(t0, 16);
      float lo = (g == 0) ? p0[r] : u;   // window key r
      float hi = (g == 0) ? u : p1[r];   // window key 8+r
      pb[r]     = (_Float16)lo;
      pb[8 + r] = (_Float16)hi;
    }

    // Out^T += V^T(d x keys) * P^T(keys x q)
    o[0] = wmma_f16(va0.v, pb, o[0]);
    o[1] = wmma_f16(va1.v, pb, o[1]);
    o[2] = wmma_f16(va2.v, pb, o[2]);
    o[3] = wmma_f16(va3.v, pb, o[3]);

    a00 = n00; a01 = n01; a10 = n10; a11 = n11;
  }

  const float inv = 1.0f / ssum;
  const int b = bh / HEADS, h = bh % HEADS;
  const int q = qbase + ln;
#pragma unroll
  for (int t = 0; t < 4; ++t)
#pragma unroll
    for (int r = 0; r < 8; ++r) {
      int d = t * 16 + r + 8 * g;
      ao[((size_t)(b * SEQ + q)) * PROJ + h * HD + d] = (_Float16)(o[t][r] * inv);
    }
}

// ---------------------------------------------------------------------------
// Output projection: (8192 x 448) @ o_w^T (448 -> 896) + o_b, f32 out.
// 32 rows x 64 cols per wave, double-buffered fragments.
// ---------------------------------------------------------------------------
__global__ __launch_bounds__(128) void oproj_kernel(
    const _Float16* __restrict__ aoh, const _Float16* __restrict__ owh,
    const float* __restrict__ ob, float* __restrict__ out)
{
  const int wv   = threadIdx.x >> 5;
  const int lane = threadIdx.x & 31;
  const int g    = lane >> 4, ln = lane & 15;
  const int mbase = blockIdx.x * 128 + wv * 32;
  const int nbase = blockIdx.y * 64;

  v8f acc[2][4] = {};
  const _Float16* arow0 = aoh + (size_t)(mbase + ln) * PROJ;
  const _Float16* arow1 = aoh + (size_t)(mbase + 16 + ln) * PROJ;
  const _Float16* wrow  = owh + (size_t)(nbase + ln) * PROJ;

  AFrag a0, a1; v16h b0, b1, b2, b3;
  a0.h[0] = *(const v8h*)(arow0 + 16 * g);
  a0.h[1] = *(const v8h*)(arow0 + 8 + 16 * g);
  a1.h[0] = *(const v8h*)(arow1 + 16 * g);
  a1.h[1] = *(const v8h*)(arow1 + 8 + 16 * g);
  b0 = *(const v16h*)(wrow + (size_t)0 * 16 * PROJ + 16 * g);
  b1 = *(const v16h*)(wrow + (size_t)1 * 16 * PROJ + 16 * g);
  b2 = *(const v16h*)(wrow + (size_t)2 * 16 * PROJ + 16 * g);
  b3 = *(const v16h*)(wrow + (size_t)3 * 16 * PROJ + 16 * g);

  for (int kk = 0; kk < PROJ; kk += 32) {
    const int kn = (kk + 32 < PROJ) ? kk + 32 : kk;
    AFrag na0, na1; v16h nb0, nb1, nb2, nb3;
    na0.h[0] = *(const v8h*)(arow0 + kn + 16 * g);
    na0.h[1] = *(const v8h*)(arow0 + kn + 8 + 16 * g);
    na1.h[0] = *(const v8h*)(arow1 + kn + 16 * g);
    na1.h[1] = *(const v8h*)(arow1 + kn + 8 + 16 * g);
    nb0 = *(const v16h*)(wrow + (size_t)0 * 16 * PROJ + kn + 16 * g);
    nb1 = *(const v16h*)(wrow + (size_t)1 * 16 * PROJ + kn + 16 * g);
    nb2 = *(const v16h*)(wrow + (size_t)2 * 16 * PROJ + kn + 16 * g);
    nb3 = *(const v16h*)(wrow + (size_t)3 * 16 * PROJ + kn + 16 * g);

    acc[0][0] = wmma_f16(a0.v, b0, acc[0][0]);
    acc[0][1] = wmma_f16(a0.v, b1, acc[0][1]);
    acc[0][2] = wmma_f16(a0.v, b2, acc[0][2]);
    acc[0][3] = wmma_f16(a0.v, b3, acc[0][3]);
    acc[1][0] = wmma_f16(a1.v, b0, acc[1][0]);
    acc[1][1] = wmma_f16(a1.v, b1, acc[1][1]);
    acc[1][2] = wmma_f16(a1.v, b2, acc[1][2]);
    acc[1][3] = wmma_f16(a1.v, b3, acc[1][3]);

    a0 = na0; a1 = na1; b0 = nb0; b1 = nb1; b2 = nb2; b3 = nb3;
  }

#pragma unroll
  for (int rr = 0; rr < 2; ++rr)
#pragma unroll
    for (int t = 0; t < 4; ++t) {
      int n = nbase + t * 16 + ln;
      float bv = ob[n];
#pragma unroll
      for (int r = 0; r < 8; ++r) {
        int m = mbase + rr * 16 + r + 8 * g;
        out[(size_t)m * HID + n] = acc[rr][t][r] + bv;
      }
    }
}

extern "C" void kernel_launch(void* const* d_in, const int* in_sizes, int n_in,
                              void* d_out, int out_size, void* d_ws, size_t ws_size,
                              hipStream_t stream) {
  (void)in_sizes; (void)n_in; (void)out_size; (void)ws_size;
  const float* x  = (const float*)d_in[0];
  const float* cs = (const float*)d_in[1];
  const float* sn = (const float*)d_in[2];
  const float* qw = (const float*)d_in[3];
  const float* qb = (const float*)d_in[4];
  const float* kw = (const float*)d_in[5];
  const float* kb = (const float*)d_in[6];
  const float* vw = (const float*)d_in[7];
  const float* vb = (const float*)d_in[8];
  const float* ow = (const float*)d_in[9];
  const float* ob = (const float*)d_in[10];
  float* out = (float*)d_out;

  const size_t NX = (size_t)BAT * SEQ * HID;       // 7,340,032
  const size_t NW = (size_t)PROJ * HID;            //   401,408
  const size_t NQ = (size_t)BAT * HEADS * SEQ * HD;// 3,670,016
  const size_t NA = (size_t)BAT * SEQ * PROJ;      // 3,670,016

  _Float16* p   = (_Float16*)d_ws;
  _Float16* xh  = p; p += NX;
  _Float16* qwh = p; p += NW;
  _Float16* kwh = p; p += NW;
  _Float16* vwh = p; p += NW;
  _Float16* owh = p; p += NW;
  _Float16* qh  = p; p += NQ;
  _Float16* kh  = p; p += NQ;
  _Float16* vth = p; p += NQ;
  _Float16* aoh = p; p += NA;

  cvt_f32_f16<<<256, 256, 0, stream>>>(x, xh, (int)NX);
  cvt_f32_f16<<<128, 256, 0, stream>>>(qw, qwh, (int)NW);
  cvt_f32_f16<<<128, 256, 0, stream>>>(kw, kwh, (int)NW);
  cvt_f32_f16<<<128, 256, 0, stream>>>(vw, vwh, (int)NW);
  cvt_f32_f16<<<128, 256, 0, stream>>>(ow, owh, (int)NW);

  qkv_rope_kernel<<<dim3(BAT * SEQ / 128, HEADS, 3), 128, 0, stream>>>(
      xh, qwh, kwh, vwh, qb, kb, vb, cs, sn, qh, kh, vth);

  attn_kernel<<<dim3(SEQ / 64, BAT * HEADS), 128, 0, stream>>>(qh, kh, vth, aoh);

  oproj_kernel<<<dim3(BAT * SEQ / 128, HID / 64), 128, 0, stream>>>(aoh, owh, ob, out);
}
